// VipCNN_35777077576246
// MI455X (gfx1250) — matrix operand order repaired and checked by reading, not compiled, not run on hardware
//
#include <hip/hip_runtime.h>
#include <stddef.h>

// ---------------- CDNA5 WMMA types ----------------
typedef __attribute__((ext_vector_type(16))) __bf16 v16bf;
typedef __attribute__((ext_vector_type(8)))  float  v8f;

union BF16x16 { v16bf v; unsigned short u[16]; };

static __device__ inline unsigned short f32_to_bf16(float f) {
  unsigned int u = __float_as_uint(f);
  unsigned int lsb = (u >> 16) & 1u;
  u += 0x7fffu + lsb;               // round-to-nearest-even
  return (unsigned short)(u >> 16);
}

#define HH 28
#define WW 28
#define BB 16
#define MTOT (BB * HH * WW)   // 12544
#define CMID 128
#define RR 6
#define DD (RR * RR)          // 36
#define NP (32 * DD)          // 1152
#define KP (256 * DD)         // 9216
#define NB (4 * DD)           // 144

// ---------------- layout/convert kernels ----------------

// NCHW f32 [16][C][28][28] -> NHWC bf16 [16][28][28][C]
__global__ void k_nchw_to_nhwc_bf16(const float* __restrict__ in,
                                    unsigned short* __restrict__ out, int C) {
  size_t i = (size_t)blockIdx.x * blockDim.x + threadIdx.x;
  size_t total = (size_t)BB * C * (HH * WW);
  if (i >= total) return;
  int c = (int)(i % C);
  size_t pix = i / C;
  int x = (int)(pix % WW);
  int y = (int)((pix / WW) % HH);
  int b = (int)(pix / (HH * WW));
  out[i] = f32_to_bf16(in[(((size_t)b * C + c) * HH + y) * WW + x]);
}

// OIHW f32 [O][I][3][3] -> [tap][O][I] bf16  (tap = ky*3+kx)
__global__ void k_prep_conv_w(const float* __restrict__ w,
                              unsigned short* __restrict__ out, int O, int I) {
  int i = blockIdx.x * blockDim.x + threadIdx.x;
  int total = 9 * O * I;
  if (i >= total) return;
  int ii = i % I;
  int o  = (i / I) % O;
  int t  = i / (I * O);
  out[i] = f32_to_bf16(w[((size_t)(o * I + ii)) * 9 + t]);
}

// ---------------- implicit-GEMM 3x3 conv (bf16 WMMA, async-to-LDS double buffer) ----
// A:  [MTOT][K] bf16 NHWC activations   (M = b*784 + y*28 + x)
// Wt: [9][128][K] bf16 (tap-major, N-major, K contiguous)
// out[gm][n] = sum_taps sum_k A[shift(gm,tap)][k] * Wt[tap][n][k] + bias[n]
// optional: + res[gm][n]; optional BN(+ReLU) epilogue.
__global__ __launch_bounds__(128) void k_conv3x3_wmma(
    const unsigned short* __restrict__ A,
    const unsigned short* __restrict__ Wt,
    const float* __restrict__ bias,
    const float* __restrict__ res,          // nullable
    const float* __restrict__ bng, const float* __restrict__ bnb,
    const float* __restrict__ bnm, const float* __restrict__ bnv, // nullable set
    float* __restrict__ outF, int K, int kstepShift) {
  __shared__ __align__(16) unsigned short Atile[2][16][32];  // double buffered, 2KB

  const int tid  = threadIdx.x;
  const int wave = tid >> 5;
  const int lane = tid & 31;
  const int m0   = blockIdx.x * 16;
  const int nhalf = lane & 15;
  const int hiK   = (lane >= 16) ? 16 : 0;
  const int n0    = wave * 32;            // this wave owns N tiles n0 and n0+16
  const int ksteps = K >> 5;
  const int total  = 9 * ksteps;

  // fixed per-thread staging coords: threads 0..63 each own one 16B chunk
  const int srow = (tid >> 2) & 15;       // A-tile row 0..15
  const int sc8  = (tid & 3) * 8;         // k-offset in shorts: 0,8,16,24
  int sb, sy0, sx0;
  {
    int gm  = m0 + srow;
    sb  = gm / (HH * WW);
    int rem = gm % (HH * WW);
    sy0 = rem / WW;
    sx0 = rem % WW;
  }

  v8f acc0 = {};
  v8f acc1 = {};

  // stage one (tap, k-step) A tile into buffer `buf` using async-to-LDS;
  // zero-fill out-of-image rows with a plain LDS store.
  auto stage = [&](int buf, int it) {
    if (tid < 64) {
      int t  = it >> kstepShift;
      int k0 = (it & (ksteps - 1)) << 5;
      int y = sy0 + t / 3 - 1;
      int x = sx0 + t % 3 - 1;
      if ((unsigned)y < (unsigned)HH && (unsigned)x < (unsigned)WW) {
        const unsigned short* src =
            A + ((size_t)((sb * HH + y) * WW + x)) * K + k0 + sc8;
        unsigned lds = (unsigned)(size_t)(const void*)&Atile[buf][srow][sc8];
        asm volatile("global_load_async_to_lds_b128 %0, %1, off"
                     :
                     : "v"(lds), "v"((unsigned long long)(size_t)src)
                     : "memory");
      } else {
        uint4 z{0u, 0u, 0u, 0u};
        *(uint4*)&Atile[buf][srow][sc8] = z;
      }
    }
  };

  stage(0, 0);
  asm volatile("s_wait_asynccnt 0x0" ::: "memory");
  __syncthreads();

  for (int it = 0; it < total; ++it) {
    const int buf = it & 1;
    if (it + 1 < total) stage(buf ^ 1, it + 1);   // overlap copy with compute

    const int t  = it >> kstepShift;
    const int k0 = (it & (ksteps - 1)) << 5;

    // A fragment (contiguous k: element s holds k = 16*(lane/16) + s, same as B)
    BF16x16 af;
    {
      int row  = lane & 15;
      int koff = (lane < 16) ? 0 : 16;
#pragma unroll
      for (int j = 0; j < 8; ++j) af.u[j] = Atile[buf][row][koff + j];
#pragma unroll
      for (int j = 0; j < 8; ++j) af.u[8 + j] = Atile[buf][row][koff + 8 + j];
    }
    // B fragments: K contiguous per lane (weights stored [tap][N][K])
    BF16x16 bf0, bf1;
    {
      const unsigned short* w0 =
          Wt + ((size_t)(t * CMID + n0 + nhalf)) * K + k0 + hiK;
      const unsigned short* w1 =
          Wt + ((size_t)(t * CMID + n0 + 16 + nhalf)) * K + k0 + hiK;
      __builtin_prefetch(w0 + 32, 0, 1);   // global_prefetch_b8
#pragma unroll
      for (int j = 0; j < 16; ++j) { bf0.u[j] = w0[j]; bf1.u[j] = w1[j]; }
    }
    acc0 = __builtin_amdgcn_wmma_f32_16x16x32_bf16(false, af.v, false, bf0.v,
                                                   (short)0, acc0, false, false);
    acc1 = __builtin_amdgcn_wmma_f32_16x16x32_bf16(false, af.v, false, bf1.v,
                                                   (short)0, acc1, false, false);

    asm volatile("s_wait_asynccnt 0x0" ::: "memory");  // next buffer fully in LDS
    __syncthreads();
  }

  // fused epilogue: bias (+res) (+BN+ReLU), C/D layout per ISA 7.12.2
#pragma unroll
  for (int r = 0; r < 8; ++r) {
    int mrow = (lane < 16) ? r : r + 8;
    size_t gm = (size_t)(m0 + mrow);
#pragma unroll
    for (int half = 0; half < 2; ++half) {
      int col = n0 + half * 16 + nhalf;
      float x = (half ? acc1[r] : acc0[r]) + bias[col];
      size_t idx = gm * CMID + col;
      if (res) x += res[idx];
      if (bng) {
        float s = bng[col] * rsqrtf(bnv[col] + 1e-5f);
        x = (x - bnm[col]) * s + bnb[col];
        x = fmaxf(x, 0.f);
      }
      outF[idx] = x;
    }
  }
}

// ---------------- BN+ReLU elementwise (also emits bf16 copy) ----------------
__global__ void k_bnrelu(const float* __restrict__ x, const float* __restrict__ g,
                         const float* __restrict__ bta, const float* __restrict__ m,
                         const float* __restrict__ v, float* __restrict__ outF,
                         unsigned short* __restrict__ outB, int total) {
  int i = blockIdx.x * blockDim.x + threadIdx.x;
  if (i >= total) return;
  int c = i & (CMID - 1);
  float s = g[c] * rsqrtf(v[c] + 1e-5f);
  float y = (x[i] - m[c]) * s + bta[c];
  y = fmaxf(y, 0.f);
  outF[i] = y;
  outB[i] = f32_to_bf16(y);
}

// ---------------- ROI pool (faithful masked-max) ----------------
__device__ inline void fix2(int& a, int& b) {
  if (b <= a) {
    if (a > 0) a -= 1; else b += 1;
  }
}

__global__ void k_roi_pool(const float* __restrict__ gA,   // channels 0..127
                           const float* __restrict__ bA,   // channels 128..255
                           const float* __restrict__ bbox_s,
                           const float* __restrict__ bbox_o, int mode,
                           float* __restrict__ out) {      // [16][9216]
  __shared__ unsigned int mrow[RR], mcol[RR];
  int b = blockIdx.x;
  if (threadIdx.x == 0) {
    const float* s = bbox_s + b * 4;
    const float* o = bbox_o + b * 4;
    float r0, r1, r2, r3;
    if (mode == 0)      { r0 = s[0]; r1 = s[1]; r2 = s[2]; r3 = s[3]; }
    else if (mode == 1) { r0 = o[0]; r1 = o[1]; r2 = o[2]; r3 = o[3]; }
    else {
      r0 = fminf(s[0], o[0]); r1 = fmaxf(s[1], o[1]);
      r2 = fminf(s[2], o[2]); r3 = fmaxf(s[3], o[3]);
    }
    int b0 = (int)floorf(r0 * (float)HH), b1 = (int)floorf(r1 * (float)HH);
    int b2 = (int)floorf(r2 * (float)WW), b3 = (int)floorf(r3 * (float)WW);
    fix2(b0, b1);
    fix2(b2, b3);
    int h = b1 - b0, w = b3 - b2;
    int mx = b2 > b3 ? b2 : b3;
    int l = RR * ((mx + RR - 1) / RR);
    if (l < RR) l = RR;                 // k >= 1 (unreachable after fix2)
    int k = l / RR;
    for (int p = 0; p < RR; ++p) { mrow[p] = 0; mcol[p] = 0; }
    for (int i = 0; i < l; ++i) {
      int p  = i / k;
      int sr = (i * h) / l + b0;
      int sc = (i * w) / l + b2;
      if ((unsigned)sr < (unsigned)HH) mrow[p] |= 1u << sr;
      if ((unsigned)sc < (unsigned)WW) mcol[p] |= 1u << sc;
    }
  }
  __syncthreads();
  for (int oi = threadIdx.x; oi < 256 * DD; oi += blockDim.x) {
    int c = oi / DD, pq = oi % DD, p = pq / RR, q = pq % RR;
    const float* src = (c < CMID) ? gA : bA;
    int cc = c & (CMID - 1);
    unsigned int Mr = mrow[p], Mc = mcol[q];
    float m = -1e30f;
    for (int y = 0; y < HH; ++y)
      if ((Mr >> y) & 1u) {
        const float* row = src + ((size_t)(b * HH * WW + y * WW)) * CMID + cc;
        for (int x = 0; x < WW; ++x)
          if ((Mc >> x) & 1u) m = fmaxf(m, row[(size_t)x * CMID]);
      }
    out[(size_t)b * (256 * DD) + oi] = m;
  }
}

// ---------------- fused multi-term linear + BN + ReLU (bf16 WMMA) ----------------
// Y[16][N] = relu(bn( sum_t A_t[16][K_t] @ W_t^T + sum_t bias_t ))
// A/W are f32 in memory; converted to bf16 in-register (each element read once).
__global__ __launch_bounds__(256) void k_linear_wmma(
    const float* __restrict__ A0, const float* __restrict__ W0,
    const float* __restrict__ Bi0, int K0,
    const float* __restrict__ A1, const float* __restrict__ W1,
    const float* __restrict__ Bi1, int K1,
    const float* __restrict__ A2, const float* __restrict__ W2,
    const float* __restrict__ Bi2, int K2, int N,
    const float* __restrict__ bng, const float* __restrict__ bnb,
    const float* __restrict__ bnm, const float* __restrict__ bnv,
    float* __restrict__ Y) {
  const int wave = threadIdx.x >> 5;
  const int lane = threadIdx.x & 31;
  const int n0 = (blockIdx.x * 8 + wave) * 16;
  if (n0 >= N) return;                       // wave-uniform exit (EXEC stays full)
  const int nhalf = lane & 15;
  const int hiK   = (lane >= 16) ? 16 : 0;
  const int arow  = lane & 15;
  const int aoff  = (lane < 16) ? 0 : 16;

  v8f acc = {};
  const float* As[3] = {A0, A1, A2};
  const float* Ws[3] = {W0, W1, W2};
  const int    Ks[3] = {K0, K1, K2};

  for (int t = 0; t < 3; ++t) {
    const float* A = As[t];
    if (!A) continue;
    const float* W = Ws[t];
    const int K = Ks[t];
    for (int k0 = 0; k0 < K; k0 += 32) {
      BF16x16 af, bf;
      const float* ap = A + (size_t)arow * K + k0;
#pragma unroll
      for (int j = 0; j < 8; ++j) af.u[j] = f32_to_bf16(ap[aoff + j]);
#pragma unroll
      for (int j = 0; j < 8; ++j) af.u[8 + j] = f32_to_bf16(ap[aoff + 8 + j]);
      const float* wp = W + (size_t)(n0 + nhalf) * K + k0 + hiK;
      __builtin_prefetch(wp + 32, 0, 1);
#pragma unroll
      for (int j = 0; j < 16; ++j) bf.u[j] = f32_to_bf16(wp[j]);
      acc = __builtin_amdgcn_wmma_f32_16x16x32_bf16(false, af.v, false, bf.v,
                                                    (short)0, acc, false, false);
    }
  }

  int col = n0 + nhalf;
  float bsum = 0.f;
  if (Bi0) bsum += Bi0[col];
  if (Bi1) bsum += Bi1[col];
  if (Bi2) bsum += Bi2[col];
  float s = bng[col] * rsqrtf(bnv[col] + 1e-5f);
#pragma unroll
  for (int r = 0; r < 8; ++r) {
    int mrow = (lane < 16) ? r : r + 8;
    float x = acc[r] + bsum;
    x = (x - bnm[col]) * s + bnb[col];
    x = fmaxf(x, 0.f);
    Y[(size_t)mrow * N + col] = x;
  }
}

// ---------------- final fc + predicate dot ----------------
__global__ void k_final(const float* __restrict__ bp, const float* __restrict__ fcw,
                        const float* __restrict__ fcb, const float* __restrict__ pred,
                        float* __restrict__ out) {
  int b = threadIdx.x;
  if (b >= BB) return;
  float s = 0.f;
  for (int n = 0; n < 9; ++n) {
    float acc = fcb[n];
    for (int k = 0; k < NB; ++k) acc += bp[b * NB + k] * fcw[n * NB + k];
    s += acc * pred[b * 9 + n];
  }
  out[b] = s;
}

// ---------------- driver ----------------
extern "C" void kernel_launch(void* const* d_in, const int* in_sizes, int n_in,
                              void* d_out, int out_size, void* d_ws, size_t ws_size,
                              hipStream_t stream) {
  (void)in_sizes; (void)n_in; (void)out_size; (void)ws_size;
  const float* feat_so = (const float*)d_in[0];
  const float* feat_p  = (const float*)d_in[1];
  const float* bbox_s  = (const float*)d_in[2];
  const float* bbox_o  = (const float*)d_in[3];
  const float* pred    = (const float*)d_in[4];
  // params in setup_inputs() insertion order
  const float* w_conv_so = (const float*)d_in[5];
  const float* b_conv_so = (const float*)d_in[6];
  const float* w_conv_p  = (const float*)d_in[7];
  const float* b_conv_p  = (const float*)d_in[8];
  const float* w_so2p    = (const float*)d_in[9];
  const float* b_so2p    = (const float*)d_in[10];
  const float* w_p2s     = (const float*)d_in[11];
  const float* b_p2s     = (const float*)d_in[12];
  const float* w_p2o     = (const float*)d_in[13];
  const float* b_p2o     = (const float*)d_in[14];
  const float* bn_gso[4] = {(const float*)d_in[15], (const float*)d_in[16],
                            (const float*)d_in[17], (const float*)d_in[18]};
  const float* bn_gp[4]  = {(const float*)d_in[19], (const float*)d_in[20],
                            (const float*)d_in[21], (const float*)d_in[22]};
  const float* bn_bp[4]  = {(const float*)d_in[23], (const float*)d_in[24],
                            (const float*)d_in[25], (const float*)d_in[26]};
  const float* bn_bs[4]  = {(const float*)d_in[27], (const float*)d_in[28],
                            (const float*)d_in[29], (const float*)d_in[30]};
  const float* bn_bo[4]  = {(const float*)d_in[31], (const float*)d_in[32],
                            (const float*)d_in[33], (const float*)d_in[34]};
  const float* w_gso = (const float*)d_in[35];
  const float* b_gso = (const float*)d_in[36];
  const float* w_gp  = (const float*)d_in[37];
  const float* b_gp  = (const float*)d_in[38];
  const float* w_s2p = (const float*)d_in[39];
  const float* b_s2p = (const float*)d_in[40];
  const float* w_o2p = (const float*)d_in[41];
  const float* b_o2p = (const float*)d_in[42];
  const float* w_bp  = (const float*)d_in[43];
  const float* b_bp  = (const float*)d_in[44];
  const float* bn2_s[4]  = {(const float*)d_in[45], (const float*)d_in[46],
                            (const float*)d_in[47], (const float*)d_in[48]};
  const float* bn2_o[4]  = {(const float*)d_in[49], (const float*)d_in[50],
                            (const float*)d_in[51], (const float*)d_in[52]};
  const float* bn2_p[4]  = {(const float*)d_in[53], (const float*)d_in[54],
                            (const float*)d_in[55], (const float*)d_in[56]};
  const float* bn2_bp[4] = {(const float*)d_in[57], (const float*)d_in[58],
                            (const float*)d_in[59], (const float*)d_in[60]};
  const float* fcw = (const float*)d_in[61];
  const float* fcb = (const float*)d_in[62];

  // workspace carve-out (256B aligned)
  char* base = (char*)d_ws;
  size_t off = 0;
  auto alloc = [&](size_t bytes) -> void* {
    void* p = base + off;
    off += (bytes + 255) & ~(size_t)255;
    return p;
  };
  unsigned short* A_so  = (unsigned short*)alloc((size_t)MTOT * 512 * 2);
  unsigned short* A_p   = (unsigned short*)alloc((size_t)MTOT * 512 * 2);
  unsigned short* Wso   = (unsigned short*)alloc((size_t)9 * 128 * 512 * 2);
  unsigned short* Wp    = (unsigned short*)alloc((size_t)9 * 128 * 512 * 2);
  unsigned short* Wso2p = (unsigned short*)alloc((size_t)9 * 128 * 128 * 2);
  unsigned short* Wp2s  = (unsigned short*)alloc((size_t)9 * 128 * 128 * 2);
  unsigned short* Wp2o  = (unsigned short*)alloc((size_t)9 * 128 * 128 * 2);
  float* f_so = (float*)alloc((size_t)MTOT * CMID * 4);
  float* f_p  = (float*)alloc((size_t)MTOT * CMID * 4);
  float* g_so = (float*)alloc((size_t)MTOT * CMID * 4);
  float* g_p  = (float*)alloc((size_t)MTOT * CMID * 4);
  float* bp2  = (float*)alloc((size_t)MTOT * CMID * 4);   // b_p
  float* b_s  = (float*)alloc((size_t)MTOT * CMID * 4);
  float* b_o  = (float*)alloc((size_t)MTOT * CMID * 4);
  unsigned short* gso_bf = (unsigned short*)alloc((size_t)MTOT * CMID * 2);
  unsigned short* bp_bf  = (unsigned short*)alloc((size_t)MTOT * CMID * 2);
  float* ps = (float*)alloc((size_t)BB * KP * 4);
  float* po = (float*)alloc((size_t)BB * KP * 4);
  float* pp = (float*)alloc((size_t)BB * KP * 4);
  float* gs = (float*)alloc((size_t)BB * NP * 4);
  float* go = (float*)alloc((size_t)BB * NP * 4);
  float* gp = (float*)alloc((size_t)BB * NP * 4);
  float* bpv = (float*)alloc((size_t)BB * NB * 4);

  const int T = 256;
  // 1) layout conversion
  {
    size_t tot = (size_t)BB * 512 * HH * WW;
    int blocks = (int)((tot + T - 1) / T);
    k_nchw_to_nhwc_bf16<<<blocks, T, 0, stream>>>(feat_so, A_so, 512);
    k_nchw_to_nhwc_bf16<<<blocks, T, 0, stream>>>(feat_p, A_p, 512);
  }
  // 2) weight prep
  k_prep_conv_w<<<(9 * 128 * 512 + T - 1) / T, T, 0, stream>>>(w_conv_so, Wso, 128, 512);
  k_prep_conv_w<<<(9 * 128 * 512 + T - 1) / T, T, 0, stream>>>(w_conv_p, Wp, 128, 512);
  k_prep_conv_w<<<(9 * 128 * 128 + T - 1) / T, T, 0, stream>>>(w_so2p, Wso2p, 128, 128);
  k_prep_conv_w<<<(9 * 128 * 128 + T - 1) / T, T, 0, stream>>>(w_p2s, Wp2s, 128, 128);
  k_prep_conv_w<<<(9 * 128 * 128 + T - 1) / T, T, 0, stream>>>(w_p2o, Wp2o, 128, 128);

  const int convGrid = MTOT / 16;  // 784
  // 3) 512->128 convs (raw f32 outputs kept for residuals)
  k_conv3x3_wmma<<<convGrid, 128, 0, stream>>>(A_so, Wso, b_conv_so, nullptr,
                                               nullptr, nullptr, nullptr, nullptr,
                                               f_so, 512, 4);
  k_conv3x3_wmma<<<convGrid, 128, 0, stream>>>(A_p, Wp, b_conv_p, nullptr,
                                               nullptr, nullptr, nullptr, nullptr,
                                               f_p, 512, 4);
  // 4) g_so = relu(bn_gso(f_so)); b_p = relu(bn_bp(f_p))  (+bf16 copies)
  {
    int tot = MTOT * CMID;
    int blocks = (tot + T - 1) / T;
    k_bnrelu<<<blocks, T, 0, stream>>>(f_so, bn_gso[0], bn_gso[1], bn_gso[2],
                                       bn_gso[3], g_so, gso_bf, tot);
    k_bnrelu<<<blocks, T, 0, stream>>>(f_p, bn_bp[0], bn_bp[1], bn_bp[2],
                                       bn_bp[3], bp2, bp_bf, tot);
  }
  // 5) fused 128->128 convs: res + bn + relu in epilogue
  k_conv3x3_wmma<<<convGrid, 128, 0, stream>>>(gso_bf, Wso2p, b_so2p, f_p,
                                               bn_gp[0], bn_gp[1], bn_gp[2], bn_gp[3],
                                               g_p, 128, 2);
  k_conv3x3_wmma<<<convGrid, 128, 0, stream>>>(bp_bf, Wp2s, b_p2s, f_so,
                                               bn_bs[0], bn_bs[1], bn_bs[2], bn_bs[3],
                                               b_s, 128, 2);
  k_conv3x3_wmma<<<convGrid, 128, 0, stream>>>(bp_bf, Wp2o, b_p2o, f_so,
                                               bn_bo[0], bn_bo[1], bn_bo[2], bn_bo[3],
                                               b_o, 128, 2);
  // 6) ROI pooling on the three concatenated feature maps
  k_roi_pool<<<BB, 256, 0, stream>>>(g_so, b_s, bbox_s, bbox_o, 0, ps);
  k_roi_pool<<<BB, 256, 0, stream>>>(g_so, b_o, bbox_s, bbox_o, 1, po);
  k_roi_pool<<<BB, 256, 0, stream>>>(g_p, bp2, bbox_s, bbox_o, 2, pp);
  // 7) linear stack (WMMA, fused BN+ReLU)
  {
    int blocksN = (NP / 16 + 7) / 8;  // 9
    k_linear_wmma<<<blocksN, 256, 0, stream>>>(
        ps, w_gso, b_gso, KP, nullptr, nullptr, nullptr, 0, nullptr, nullptr,
        nullptr, 0, NP, bn2_s[0], bn2_s[1], bn2_s[2], bn2_s[3], gs);
    k_linear_wmma<<<blocksN, 256, 0, stream>>>(
        po, w_gso, b_gso, KP, nullptr, nullptr, nullptr, 0, nullptr, nullptr,
        nullptr, 0, NP, bn2_o[0], bn2_o[1], bn2_o[2], bn2_o[3], go);
    k_linear_wmma<<<blocksN, 256, 0, stream>>>(
        pp, w_gp, b_gp, KP, gs, w_s2p, b_s2p, NP, go, w_o2p, b_o2p, NP, NP,
        bn2_p[0], bn2_p[1], bn2_p[2], bn2_p[3], gp);
    int blocksB = (NB / 16 + 7) / 8;  // 2
    k_linear_wmma<<<blocksB, 256, 0, stream>>>(
        gp, w_bp, b_bp, NP, nullptr, nullptr, nullptr, 0, nullptr, nullptr,
        nullptr, 0, NB, bn2_bp[0], bn2_bp[1], bn2_bp[2], bn2_bp[3], bpv);
  }
  // 8) logits + predicate reduce -> out[16]
  k_final<<<1, 32, 0, stream>>>(bpv, fcw, fcb, pred, (float*)d_out);
}
